// MultiHeadAttention_57715770523709
// MI455X (gfx1250) — matrix operand order repaired and checked by reading, not compiled, not run on hardware
//
#include <hip/hip_runtime.h>

// ---------------------------------------------------------------------------
// MI455X / gfx1250 multi-head attention (faithful softmax-over-heads variant)
// B=2, F=2048, D=1024, H=16, DK=64
// d_out = [ out (B*F*D fp32) | attn (B*H*F*F fp32) ]
// Memory-bound on attn (537MB write + 537MB read) -> bf16 WMMA everywhere,
// fp32 accumulate, single-pass head-softmax fused in registers.
// ---------------------------------------------------------------------------

#define Bc 2
#define Fc 2048
#define Dc 1024
#define Hc 16
#define DKc 64

typedef __attribute__((ext_vector_type(16))) __bf16        v16bf;
typedef __attribute__((ext_vector_type(8)))  float         v8f;
typedef __attribute__((ext_vector_type(4)))  unsigned int  u32x4;
typedef __attribute__((ext_vector_type(4)))  float         f32x4;

union FragU {
    v16bf v;
    u32x4 q[2];
    unsigned short us[16];
};

__device__ __forceinline__ unsigned short f2bf(float f) {
    unsigned u = __builtin_bit_cast(unsigned, f);
    unsigned r = u + 0x7FFFu + ((u >> 16) & 1u);   // round-to-nearest-even
    return (unsigned short)(r >> 16);
}

__device__ __forceinline__ v8f wmma_bf16(const FragU& a, const FragU& b, v8f c) {
    return __builtin_amdgcn_wmma_f32_16x16x32_bf16(
        /*neg_a=*/false, a.v, /*neg_b=*/false, b.v,
        /*c_mod=*/(short)0, c, /*reuse_a=*/false, /*reuse_b=*/false);
}

// ------------------------------ fp32 -> bf16 -------------------------------
__global__ void cvt_f32_bf16(const float* __restrict__ in,
                             unsigned short* __restrict__ out, int n) {
    int i = blockIdx.x * blockDim.x + threadIdx.x;
    if (i < n) out[i] = f2bf(in[i]);
}

// ------------------------- tiled bf16 GEMM (+bias) -------------------------
// C[M=4096, N=1024] = A[4096x1024] @ W^T + bias
//   mode 0: store bf16 head-major qp/kp/vp [B,H,F,DK]
//   mode 1: store fp32 flat [M,1024] (final projection)
// block 128 = 4 waves; block tile 64x64; wave tile 16x64 (4 WMMA accums)
__global__ __launch_bounds__(128) void gemm_bf16(
    const unsigned short* __restrict__ A,   // bf16 raw
    const unsigned short* __restrict__ W,   // bf16 raw, row = out feature
    const float* __restrict__ bias,
    void* __restrict__ Cout,
    int mode)
{
    __shared__ unsigned short As[64][48];   // [m][k], 96B row stride (16B aligned)
    __shared__ unsigned short Bs[64][48];   // [n][k]

    const int t     = threadIdx.x;
    const int w     = t >> 5;
    const int l     = t & 31;
    const int khalf = l >> 4;
    const int ln    = l & 15;
    const int row0  = blockIdx.x * 64;
    const int col0  = blockIdx.y * 64;
    const int base0 = khalf * 8;

    v8f z = {0.f,0.f,0.f,0.f,0.f,0.f,0.f,0.f};
    v8f acc[4] = {z, z, z, z};

    const int lr = t >> 1;             // 0..63
    const int lc = (t & 1) * 16;       // 0 or 16

    for (int kk = 0; kk < Dc; kk += 32) {
        // stage A tile (64x32) and W tile (Bs[n][k] = W[col0+n][kk+k])
        const unsigned short* asrc = A + (size_t)(row0 + lr) * Dc + kk + lc;
        const unsigned short* wsrc = W + (size_t)(col0 + lr) * Dc + kk + lc;
        __builtin_prefetch(asrc + 32, 0, 0);   // next k-tile -> global_prefetch_b8
        __builtin_prefetch(wsrc + 32, 0, 0);
        *(u32x4*)&As[lr][lc]     = *(const u32x4*)(asrc);
        *(u32x4*)&As[lr][lc + 8] = *(const u32x4*)(asrc + 8);
        *(u32x4*)&Bs[lr][lc]     = *(const u32x4*)(wsrc);
        *(u32x4*)&Bs[lr][lc + 8] = *(const u32x4*)(wsrc + 8);
        __syncthreads();

        FragU af;
        const int arow = w * 16 + ln;
        af.q[0] = *(const u32x4*)&As[arow][base0];        // K = base0..base0+7
        af.q[1] = *(const u32x4*)&As[arow][16 + base0];   // K = 16+base0..+7
#pragma unroll
        for (int nt = 0; nt < 4; ++nt) {
            FragU bfg;
            const int ncol = nt * 16 + ln;
            bfg.q[0] = *(const u32x4*)&Bs[ncol][base0];      // K = base0..base0+7
            bfg.q[1] = *(const u32x4*)&Bs[ncol][16 + base0];
            acc[nt] = wmma_bf16(af, bfg, acc[nt]);
        }
        __syncthreads();
    }

#pragma unroll
    for (int nt = 0; nt < 4; ++nt) {
        const int c  = col0 + nt * 16 + ln;
        const float bv = bias[c];
#pragma unroll
        for (int r = 0; r < 8; ++r) {
            const int m  = row0 + w * 16 + r + khalf * 8;
            const float val = acc[nt][r] + bv;
            if (mode == 0) {
                const int b = m >> 11, f = m & (Fc - 1);
                const int h = c >> 6,  dk = c & (DKc - 1);
                ((unsigned short*)Cout)[(((size_t)(b * Hc + h) * Fc + f) * DKc) + dk] = f2bf(val);
            } else {
                ((float*)Cout)[(size_t)m * Dc + c] = val;
            }
        }
    }
}

// ------------------ scores + softmax over HEADS (axis=1) -------------------
// Each wave: one 16(q) x 16(k) tile for ALL 16 heads, kept in registers.
// attn[b,h,q,k] = exp(s/32) / sum_h exp(s/32)
__global__ __launch_bounds__(128) void scores_softmax_h(
    const unsigned short* __restrict__ qp,   // bf16 [B,H,F,DK]
    const unsigned short* __restrict__ kp,   // bf16 [B,H,F,DK]
    float* __restrict__ attn)                // fp32 [B,H,F,F]
{
    const int t     = threadIdx.x;
    const int w     = t >> 5;
    const int l     = t & 31;
    const int khalf = l >> 4;
    const int ln    = l & 15;

    const int bid = blockIdx.x;             // B * (F/16) * (F/64) = 8192
    const int b   = bid >> 12;
    const int r   = bid & 4095;
    const int qt  = r >> 5;
    const int kg  = r & 31;
    const int q0  = qt * 16;
    const int k0  = kg * 64 + w * 16;
    const int base0 = khalf * 8;

    v8f z = {0.f,0.f,0.f,0.f,0.f,0.f,0.f,0.f};
    v8f acc[16];

#pragma unroll
    for (int h = 0; h < Hc; ++h) {
        v8f c = z;
        const unsigned short* qrow = qp + ((size_t)((b * Hc + h) * Fc) + q0 + ln) * DKc;
        const unsigned short* krow = kp + ((size_t)((b * Hc + h) * Fc) + k0 + ln) * DKc;
#pragma unroll
        for (int db = 0; db < DKc; db += 32) {
            FragU af, bfg;
            af.q[0]  = *(const u32x4*)(qrow + db + base0);
            af.q[1]  = *(const u32x4*)(qrow + db + 16 + base0);
            bfg.q[0] = *(const u32x4*)(krow + db + base0);
            bfg.q[1] = *(const u32x4*)(krow + db + 16 + base0);
            c = wmma_bf16(af, bfg, c);
        }
        acc[h] = c;
    }

    float sum[8] = {0.f,0.f,0.f,0.f,0.f,0.f,0.f,0.f};
#pragma unroll
    for (int h = 0; h < Hc; ++h)
#pragma unroll
        for (int i = 0; i < 8; ++i) {
            const float e = __expf(acc[h][i] * 0.03125f);   // 1/sqrt(1024)
            acc[h][i] = e;
            sum[i] += e;
        }
#pragma unroll
    for (int i = 0; i < 8; ++i) sum[i] = 1.f / sum[i];

#pragma unroll
    for (int h = 0; h < Hc; ++h) {
        float* orow = attn + (size_t)((b * Hc + h) * Fc) * Fc;
#pragma unroll
        for (int i = 0; i < 8; ++i) {
            const int qr = q0 + i + khalf * 8;
            orow[(size_t)qr * Fc + k0 + ln] = acc[h][i] * sum[i];
        }
    }
}

// ----------------- attn @ V, stored pre-scrambled (bf16) -------------------
// res[b,h,f,d] -> scrambled[b'=h/8, f'=(h%8)*256+f/8, d'=(f%8)*128+b*64+d]
__global__ __launch_bounds__(128) void attnv_scramble(
    const float* __restrict__ attn,          // fp32 [B,H,F,F]
    const unsigned short* __restrict__ vp,   // bf16 [B,H,F,DK]
    unsigned short* __restrict__ resb)       // bf16 [4096,1024] scrambled
{
    __shared__ unsigned short Vt[64][48];    // Vt[d][k] (transposed V tile)

    const int t     = threadIdx.x;
    const int w     = t >> 5;
    const int l     = t & 31;
    const int khalf = l >> 4;
    const int ln    = l & 15;

    const int bid = blockIdx.x;              // B*H*(F/64) = 1024
    const int qt  = bid & 31;
    const int bh  = bid >> 5;                // b*16 + h
    const int h   = bh & (Hc - 1);
    const int b   = bh >> 4;
    const int q0  = qt * 64 + w * 16;
    const int base0 = khalf * 8;

    v8f z = {0.f,0.f,0.f,0.f,0.f,0.f,0.f,0.f};
    v8f acc[4] = {z, z, z, z};

    const float* arow = attn + ((size_t)(bh * Fc) + q0 + ln) * Fc;
    const int kr = t & 31;
    const int dc = (t >> 5) * 16;

    for (int kk = 0; kk < Fc; kk += 32) {
        // stage Vt[d][k] = vp[bh][kk+k][d]
        {
            const unsigned short* vrow = vp + ((size_t)(bh * Fc) + kk + kr) * DKc + dc;
            unsigned short tmp[16];
            *(u32x4*)tmp       = *(const u32x4*)(vrow);
            *(u32x4*)(tmp + 8) = *(const u32x4*)(vrow + 8);
#pragma unroll
            for (int i = 0; i < 16; ++i) Vt[dc + i][kr] = tmp[i];
        }
        __syncthreads();

        // A fragment: attn fp32 -> bf16
        FragU af;
        f32x4 a0 = *(const f32x4*)(arow + kk + base0);
        f32x4 a1 = *(const f32x4*)(arow + kk + base0 + 4);
        f32x4 a2 = *(const f32x4*)(arow + kk + 16 + base0);
        f32x4 a3 = *(const f32x4*)(arow + kk + 16 + base0 + 4);
#pragma unroll
        for (int i = 0; i < 4; ++i) {
            af.us[i]      = f2bf(a0[i]);
            af.us[4 + i]  = f2bf(a1[i]);
            af.us[8 + i]  = f2bf(a2[i]);
            af.us[12 + i] = f2bf(a3[i]);
        }
#pragma unroll
        for (int nt = 0; nt < 4; ++nt) {
            FragU bfg;
            bfg.q[0] = *(const u32x4*)&Vt[nt * 16 + ln][base0];
            bfg.q[1] = *(const u32x4*)&Vt[nt * 16 + ln][16 + base0];
            acc[nt] = wmma_bf16(af, bfg, acc[nt]);
        }
        __syncthreads();
    }

#pragma unroll
    for (int nt = 0; nt < 4; ++nt) {
#pragma unroll
        for (int i = 0; i < 8; ++i) {
            const int f  = q0 + i + khalf * 8;
            const int d  = nt * 16 + ln;
            const int bp = h >> 3;
            const int fp = (h & 7) * 256 + (f >> 3);
            const int dp = (f & 7) * 128 + b * 64 + d;
            resb[((size_t)(bp * Fc) + fp) * Dc + dp] = f2bf(acc[nt][i]);
        }
    }
}

// ---------------------------------------------------------------------------
extern "C" void kernel_launch(void* const* d_in, const int* in_sizes, int n_in,
                              void* d_out, int out_size, void* d_ws, size_t ws_size,
                              hipStream_t stream) {
    // input order: k, q, v, Wq, bq, Wk, bk, Wv, bv, W2, b2
    const float* k_in = (const float*)d_in[0];
    const float* q_in = (const float*)d_in[1];
    const float* v_in = (const float*)d_in[2];
    const float* Wq   = (const float*)d_in[3];
    const float* bq   = (const float*)d_in[4];
    const float* Wk   = (const float*)d_in[5];
    const float* bk   = (const float*)d_in[6];
    const float* Wv   = (const float*)d_in[7];
    const float* bv   = (const float*)d_in[8];
    const float* W2   = (const float*)d_in[9];
    const float* b2   = (const float*)d_in[10];

    const int NTOK = Bc * Fc;          // 4096
    const int NX   = NTOK * Dc;        // 4,194,304  elements (q/k/v, proj)
    const int NW   = Dc * Dc;          // 1,048,576  elements (weights)

    // workspace carve-up (bf16 = ushort), ~64 MB total
    unsigned short* qb   = (unsigned short*)d_ws;
    unsigned short* kb   = qb + (size_t)NX;
    unsigned short* vb   = kb + (size_t)NX;
    unsigned short* Wqb  = vb + (size_t)NX;
    unsigned short* Wkb  = Wqb + (size_t)NW;
    unsigned short* Wvb  = Wkb + (size_t)NW;
    unsigned short* W2b  = Wvb + (size_t)NW;
    unsigned short* qp   = W2b + (size_t)NW;   // [B,H,F,DK]
    unsigned short* kp   = qp + (size_t)NX;
    unsigned short* vp   = kp + (size_t)NX;
    unsigned short* resb = vp + (size_t)NX;    // scrambled [4096,1024]

    float* out_p  = (float*)d_out;                       // [B,F,D]
    float* attn_p = out_p + (size_t)NX;                  // [B,H,F,F]

    // 1) fp32 -> bf16 conversions
    cvt_f32_bf16<<<(NX + 255) / 256, 256, 0, stream>>>(q_in, qb, NX);
    cvt_f32_bf16<<<(NX + 255) / 256, 256, 0, stream>>>(k_in, kb, NX);
    cvt_f32_bf16<<<(NX + 255) / 256, 256, 0, stream>>>(v_in, vb, NX);
    cvt_f32_bf16<<<(NW + 255) / 256, 256, 0, stream>>>(Wq, Wqb, NW);
    cvt_f32_bf16<<<(NW + 255) / 256, 256, 0, stream>>>(Wk, Wkb, NW);
    cvt_f32_bf16<<<(NW + 255) / 256, 256, 0, stream>>>(Wv, Wvb, NW);
    cvt_f32_bf16<<<(NW + 255) / 256, 256, 0, stream>>>(W2, W2b, NW);

    // 2) QKV projections -> head-major bf16 (WMMA GEMMs)
    dim3 ggrid(NTOK / 64, Dc / 64);    // (64, 16)
    gemm_bf16<<<ggrid, 128, 0, stream>>>(qb, Wqb, bq, (void*)qp, 0);
    gemm_bf16<<<ggrid, 128, 0, stream>>>(kb, Wkb, bk, (void*)kp, 0);
    gemm_bf16<<<ggrid, 128, 0, stream>>>(vb, Wvb, bv, (void*)vp, 0);

    // 3) scores + softmax-over-heads -> attn (second output)
    scores_softmax_h<<<Bc * (Fc / 16) * (Fc / 64), 128, 0, stream>>>(qp, kp, attn_p);

    // 4) attn @ V with faithful permute(1,2,0,3).view scramble
    attnv_scramble<<<Bc * Hc * (Fc / 64), 128, 0, stream>>>(attn_p, vp, resb);

    // 5) final projection: out = scrambled_res @ W2^T + b2 (fp32 out)
    gemm_bf16<<<ggrid, 128, 0, stream>>>(resb, W2b, b2, (void*)out_p, 1);
}
